// RNN_85117661872556
// MI455X (gfx1250) — matrix operand order repaired and checked by reading, not compiled, not run on hardware
//
#include <hip/hip_runtime.h>
#include <hip/hip_bf16.h>

// GRU scan, MI455X (gfx1250, wave32, WMMA).
// Phase 1: xg = x @ Wi + bi       (big bf16-WMMA GEMM, fp32 accum, fp32 out)
// Phase 2: 16 persistent WGs run the 1024-step recurrence; Wh slice lives in
//          LDS (pre-packed in WMMA B-fragment order); h ping-pongs in bf16;
//          next step's xg slice is async-DMA'd into LDS during the GEMM.
//
// Workspace layout (needs ~387 MB):
//   [0x000000) Wi packed bf16 B-fragments   (768 KB)
//   [0x100000) Wh packed bf16 B-fragments   (1.5 MB)
//   [0x280000) h ping buffer bf16           (64 KB)
//   [0x290000) h pong buffer bf16           (64 KB)
//   [0x2A0000) barrier counter
//   [0x300000) xg fp32 (T*B, 3H)            (384 MB)

typedef __bf16 bf16;
typedef __attribute__((ext_vector_type(16))) bf16  v16bf;
typedef __attribute__((ext_vector_type(8)))  bf16  v8bf;
typedef __attribute__((ext_vector_type(8)))  float v8f;
typedef int v4i __attribute__((vector_size(16)));

#define T_DIM 1024
#define B_DIM 64
#define F_DIM 256
#define H_DIM 512
#define G_DIM (3 * H_DIM)        // 1536
#define M_DIM (T_DIM * B_DIM)    // 65536

#define NT_G (G_DIM / 16)        // 96 column tiles
#define KT_F (F_DIM / 32)        // 8
#define KT_H (H_DIM / 32)        // 16
#define NWG  16                  // persistent workgroups in the scan

#define WI_PACK_OFF 0x000000ull
#define WH_PACK_OFF 0x100000ull
#define H0_BUF_OFF  0x280000ull
#define H1_BUF_OFF  0x290000ull
#define BAR_OFF     0x2A0000ull
#define XG_OFF      0x300000ull

// LDS: 96KB packed Wh + 2 x 24KB xg ping-pong.
#define SWH_ELEMS   (96 * 512)           // bf16
#define SXG_FLOATS  (B_DIM * 96)         // 6144 floats per buffer
#define SCAN_LDS    (96 * 1024 + 2 * SXG_FLOATS * 4)   // 147456 bytes

// Async global->LDS path (gfx1250). Fall back to direct loads + prefetch.
#if defined(__gfx1250__) && \
    __has_builtin(__builtin_amdgcn_global_load_async_to_lds_b128) && \
    __has_builtin(__builtin_amdgcn_s_wait_asynccnt)
#define XG_ASYNC 1
#else
#define XG_ASYNC 0
#endif

#if XG_ASYNC
__device__ __forceinline__ void async_copy16(const float* gp, float* lp) {
    __builtin_amdgcn_global_load_async_to_lds_b128(
        (__attribute__((address_space(1))) v4i*)gp,
        (__attribute__((address_space(3))) v4i*)lp, 0, 0);
}
#endif

__device__ __forceinline__ float sigmoid_f(float x) {
    return 1.0f / (1.0f + __expf(-x));
}
__device__ __forceinline__ float tanh_f(float x) {
    return 2.0f * sigmoid_f(2.0f * x) - 1.0f;
}

// ---------------------------------------------------------------------------
// Pack a fp32 (K, N) row-major weight matrix into bf16 WMMA B-fragments.
// Contiguous-k order used here (16-bit B 32x16, A loads match): lane L holds column N = L%16, K values
// (L/16)*16 + 0..15, two packed per VGPR -> 16 contiguous bf16 per lane.
// ---------------------------------------------------------------------------
__global__ void pack_b_frags(const float* __restrict__ W, bf16* __restrict__ out,
                             int K, int N) {
    int gid = blockIdx.x * blockDim.x + threadIdx.x;
    int ntiles = (K / 32) * (N / 16);
    if (gid >= ntiles * 32) return;
    int tile = gid >> 5;
    int lane = gid & 31;
    int nt   = tile % (N / 16);
    int kt   = tile / (N / 16);
    int col   = nt * 16 + (lane & 15);
    int kbase = kt * 32 + (lane >> 4) * 16;
    bf16* dst = out + ((size_t)tile * 32 + lane) * 16;
#pragma unroll
    for (int e = 0; e < 16; ++e)
        dst[e] = (bf16)W[(size_t)(kbase + e) * N + col];
}

// Initialize bf16 h ping buffer from fp32 h0, and zero the grid barrier.
__global__ void init_scan(const float* __restrict__ h0, bf16* __restrict__ hbuf0,
                          unsigned* __restrict__ bar) {
    int i = blockIdx.x * blockDim.x + threadIdx.x;
    if (i < B_DIM * H_DIM) hbuf0[i] = (bf16)h0[i];
    if (i == 0) *bar = 0u;
}

// ---------------------------------------------------------------------------
// Phase 1: xg = x @ Wi + bi.  M=65536, K=256, N=1536, fp32 out.
// Block = 8 waves (256 thr): 2 wave-rows x 4 wave-cols -> 64 x 256 tile.
// ---------------------------------------------------------------------------
__global__ void __launch_bounds__(256) xg_gemm(const float* __restrict__ x,
                                               const float* __restrict__ bi,
                                               const bf16*  __restrict__ wiPack,
                                               float* __restrict__ xg) {
    const int lane  = threadIdx.x & 31;
    const int w     = threadIdx.x >> 5;
    const int wrow  = w & 1;
    const int wcol  = w >> 1;
    const int MT    = blockIdx.y * 64  + wrow * 32;
    const int NT    = blockIdx.x * 256 + wcol * 64;
    const int laneR = lane & 15;
    const int laneH = lane >> 4;

    const v8f zero8 = {0.f, 0.f, 0.f, 0.f, 0.f, 0.f, 0.f, 0.f};
    v8f acc[2][4];
#pragma unroll
    for (int mi = 0; mi < 2; ++mi)
#pragma unroll
        for (int ni = 0; ni < 4; ++ni) acc[mi][ni] = zero8;

    for (int kt = 0; kt < KT_F; ++kt) {
        v16bf a[2];
#pragma unroll
        for (int mi = 0; mi < 2; ++mi) {
            const float* p = x + (size_t)(MT + mi * 16 + laneR) * F_DIM
                               + kt * 32 + laneH * 16;
#pragma unroll
            for (int e = 0; e < 8; ++e) {
                a[mi][e]     = (bf16)p[e];
                a[mi][e + 8] = (bf16)p[e + 8];
            }
        }
#pragma unroll
        for (int ni = 0; ni < 4; ++ni) {
            int tile = kt * NT_G + (NT / 16 + ni);
            v16bf b = *(const v16bf*)(wiPack + ((size_t)tile * 32 + lane) * 16);
#pragma unroll
            for (int mi = 0; mi < 2; ++mi)
                acc[mi][ni] = __builtin_amdgcn_wmma_f32_16x16x32_bf16(
                    false, a[mi], false, b, (short)0, acc[mi][ni], false, false);
        }
    }

#pragma unroll
    for (int ni = 0; ni < 4; ++ni) {
        int col  = NT + ni * 16 + laneR;
        float bv = bi[col];
#pragma unroll
        for (int mi = 0; mi < 2; ++mi) {
            int row0 = MT + mi * 16 + laneH * 8;
#pragma unroll
            for (int v = 0; v < 8; ++v)
                xg[(size_t)(row0 + v) * G_DIM + col] = acc[mi][ni][v] + bv;
        }
    }
}

// ---------------------------------------------------------------------------
// Phase 2: persistent GRU scan. 16 WGs x 128 threads (4 waves).
// WG owns hidden cols [b*32, b*32+32); wave mt owns batch rows [mt*16,+16).
// ---------------------------------------------------------------------------
__global__ void __launch_bounds__(128) gru_scan(const float* __restrict__ xg,
                                                const float* __restrict__ h0,
                                                const float* __restrict__ bhn,
                                                const bf16*  __restrict__ whPack,
                                                bf16* __restrict__ hbuf0,
                                                bf16* __restrict__ hbuf1,
                                                unsigned* __restrict__ bar,
                                                float* __restrict__ out) {
    extern __shared__ char smem_raw[];
    bf16*  sWh = (bf16*)smem_raw;                       // 96 tiles * 512 elems
    float* sXg = (float*)(smem_raw + 96 * 1024);        // 2 * 6144 floats

    const int lane  = threadIdx.x & 31;
    const int mt    = threadIdx.x >> 5;
    const int laneR = lane & 15;
    const int laneH = lane >> 4;
    const int j0    = blockIdx.x * 32;
    const int jt0   = j0 >> 4;

    // Stage this WG's Wh slice into LDS, local tile lt = kt*6 + (g*2 + nt2).
    for (int c = threadIdx.x; c < 96 * 64; c += blockDim.x) {   // 16B chunks
        int lt  = c >> 6;
        int off = c & 63;
        int kt  = lt / 6, l = lt % 6;
        int g   = l >> 1, nt2 = l & 1;
        int gtile = kt * NT_G + g * (H_DIM / 16) + jt0 + nt2;
        *(int4*)(sWh + (size_t)lt * 512 + off * 8) =
            *(const int4*)(whPack + (size_t)gtile * 512 + off * 8);
    }
    __syncthreads();

    // h_old slice in registers (same C-fragment layout as the accumulators).
    float hold[2][8];
#pragma unroll
    for (int nt2 = 0; nt2 < 2; ++nt2)
#pragma unroll
        for (int v = 0; v < 8; ++v)
            hold[nt2][v] = h0[(size_t)(mt * 16 + laneH * 8 + v) * H_DIM
                              + j0 + nt2 * 16 + laneR];

    bf16* hbufs[2] = {hbuf0, hbuf1};
    const v8f zero8 = {0.f, 0.f, 0.f, 0.f, 0.f, 0.f, 0.f, 0.f};

#if XG_ASYNC
    // Async-prefetch xg[0] into LDS buffer 0 (12 instructions per wave;
    // each lane moves one 16B chunk per instruction).
    for (int i = 0; i < 12; ++i) {
        int c2   = threadIdx.x + i * 128;
        int brow = c2 / 24, rem = c2 % 24;
        int g    = rem >> 3;
        int off4 = (rem & 7) * 4;
        async_copy16(xg + (size_t)brow * G_DIM + g * H_DIM + j0 + off4,
                     sXg + brow * 96 + g * 32 + off4);
    }
#endif

    for (int t = 0; t < T_DIM; ++t) {
        const bf16* hin  = hbufs[t & 1];
        bf16*       hout = hbufs[(t + 1) & 1];

#if XG_ASYNC
        // Kick off async DMA of xg[t+1] into the other LDS buffer; it
        // overlaps the 96 WMMAs below and is waited on before the gate math.
        if (t + 1 < T_DIM) {
            const float* base = xg + (size_t)(t + 1) * B_DIM * G_DIM;
            float* dst = sXg + ((t + 1) & 1) * SXG_FLOATS;
            for (int i = 0; i < 12; ++i) {
                int c2   = threadIdx.x + i * 128;
                int brow = c2 / 24, rem = c2 % 24;
                int g    = rem >> 3;
                int off4 = (rem & 7) * 4;
                async_copy16(base + (size_t)brow * G_DIM + g * H_DIM + j0 + off4,
                             dst + brow * 96 + g * 32 + off4);
            }
        }
#else
        // Fallback: software prefetch of next step's xg slice into caches.
        if (t + 1 < T_DIM) {
            const float* base = xg + (size_t)(t + 1) * B_DIM * G_DIM;
            for (int i = 0; i < 12; ++i) {
                int c2   = threadIdx.x + i * 128;
                int brow = c2 / 24, rem = c2 % 24;
                int g    = rem >> 3;
                int off4 = (rem & 7) * 4;
                __builtin_prefetch(base + (size_t)brow * G_DIM + g * H_DIM + j0 + off4,
                                   0, 1);
            }
        }
#endif

        v8f acc[3][2];
#pragma unroll
        for (int g = 0; g < 3; ++g)
#pragma unroll
            for (int n = 0; n < 2; ++n) acc[g][n] = zero8;

        for (int kt = 0; kt < KT_H; ++kt) {
            const bf16* p = hin + (size_t)(mt * 16 + laneR) * H_DIM
                                + kt * 32 + laneH * 16;
            v8bf lo = *(const v8bf*)p;
            v8bf hi = *(const v8bf*)(p + 8);
            v16bf a;
#pragma unroll
            for (int e = 0; e < 8; ++e) { a[e] = lo[e]; a[e + 8] = hi[e]; }

#pragma unroll
            for (int g = 0; g < 3; ++g)
#pragma unroll
                for (int nt2 = 0; nt2 < 2; ++nt2) {
                    int lt = kt * 6 + g * 2 + nt2;
                    v16bf b = *(const v16bf*)(sWh + ((size_t)lt * 32 + lane) * 16);
                    acc[g][nt2] = __builtin_amdgcn_wmma_f32_16x16x32_bf16(
                        false, a, false, b, (short)0, acc[g][nt2], false, false);
                }
        }

#if XG_ASYNC
        // xg[t]'s 12 async loads (per wave) were issued last step; async loads
        // complete in order, so waiting ASYNCcnt<=12 retires them while the
        // 12 just issued for t+1 may stay in flight.
        __builtin_amdgcn_s_wait_asynccnt(12);
        __syncthreads();   // chunks were written by all waves of the WG
        const float* sx = sXg + (t & 1) * SXG_FLOATS;
#else
        const float* xgt = xg + (size_t)t * B_DIM * G_DIM;
#endif

        // GRU gate math + stores.
#pragma unroll
        for (int nt2 = 0; nt2 < 2; ++nt2) {
            int colh = j0 + nt2 * 16 + laneR;
            float bn = bhn[colh];
#pragma unroll
            for (int v = 0; v < 8; ++v) {
                int brow = mt * 16 + laneH * 8 + v;
#if XG_ASYNC
                const float* xrow = sx + brow * 96;
                float xr = xrow[nt2 * 16 + laneR];
                float xz = xrow[32 + nt2 * 16 + laneR];
                float xn = xrow[64 + nt2 * 16 + laneR];
#else
                const float* xrow = xgt + (size_t)brow * G_DIM;
                float xr = xrow[colh];
                float xz = xrow[H_DIM + colh];
                float xn = xrow[2 * H_DIM + colh];
#endif
                float r  = sigmoid_f(xr + acc[0][nt2][v]);
                float z  = sigmoid_f(xz + acc[1][nt2][v]);
                float nn = tanh_f(xn + r * (acc[2][nt2][v] + bn));
                float hv = (1.0f - z) * nn + z * hold[nt2][v];
                hold[nt2][v] = hv;
                out[(size_t)t * B_DIM * H_DIM + (size_t)brow * H_DIM + colh] = hv;
                hout[(size_t)brow * H_DIM + colh] = (bf16)hv;
            }
        }

        // Device-scope grid barrier (monotonic counter; reset each launch).
        __syncthreads();
        if (threadIdx.x == 0) {
            __threadfence();                         // release h stores
            atomicAdd(bar, 1u);
            const unsigned target = (unsigned)(t + 1) * (unsigned)NWG;
            while (__hip_atomic_load(bar, __ATOMIC_RELAXED,
                                     __HIP_MEMORY_SCOPE_AGENT) < target)
                __builtin_amdgcn_s_sleep(1);
        }
        __syncthreads();
        __threadfence();                             // acquire ordering
        // Drop any stale lines of the peer-written h buffer from this WGP's
        // near caches before the next step's A-fragment loads (ISA 7.2:
        // acquire/INV; GLOBAL_INV is tracked with LOADcnt).
        asm volatile("global_inv scope:SCOPE_DEV\n\t"
                     "s_wait_loadcnt 0x0" ::: "memory");
    }
}

extern "C" void kernel_launch(void* const* d_in, const int* in_sizes, int n_in,
                              void* d_out, int out_size, void* d_ws, size_t ws_size,
                              hipStream_t stream) {
    const float* x   = (const float*)d_in[0];   // (T, B, F)
    const float* h0  = (const float*)d_in[1];   // (B, H)
    const float* Wi  = (const float*)d_in[2];   // (F, 3H)
    const float* bi  = (const float*)d_in[3];   // (3H,)
    const float* Wh  = (const float*)d_in[4];   // (H, 3H)
    const float* bhn = (const float*)d_in[5];   // (H,)
    float* out = (float*)d_out;                 // (T, B, H)

    char* ws = (char*)d_ws;
    bf16*     wiPack = (bf16*)(ws + WI_PACK_OFF);
    bf16*     whPack = (bf16*)(ws + WH_PACK_OFF);
    bf16*     hbuf0  = (bf16*)(ws + H0_BUF_OFF);
    bf16*     hbuf1  = (bf16*)(ws + H1_BUF_OFF);
    unsigned* bar    = (unsigned*)(ws + BAR_OFF);
    float*    xgws   = (float*)(ws + XG_OFF);

    {   // pack Wi into B fragments
        int tot = KT_F * NT_G * 32;
        pack_b_frags<<<(tot + 255) / 256, 256, 0, stream>>>(Wi, wiPack, F_DIM, G_DIM);
    }
    {   // pack Wh into B fragments
        int tot = KT_H * NT_G * 32;
        pack_b_frags<<<(tot + 255) / 256, 256, 0, stream>>>(Wh, whPack, H_DIM, G_DIM);
    }
    init_scan<<<(B_DIM * H_DIM + 255) / 256, 256, 0, stream>>>(h0, hbuf0, bar);

    dim3 g1(G_DIM / 256, M_DIM / 64);           // (6, 1024)
    xg_gemm<<<g1, 256, 0, stream>>>(x, bi, wiPack, xgws);

    gru_scan<<<NWG, 128, SCAN_LDS, stream>>>(xgws, h0, bhn, whPack,
                                             hbuf0, hbuf1, bar, out);
}
